// AttnBlock_2388001817205
// MI455X (gfx1250) — matrix operand order repaired and checked by reading, not compiled, not run on hardware
//
#include <hip/hip_runtime.h>
#include <hip/hip_bf16.h>
#include <stdint.h>

// ---------------------------------------------------------------------------
// AttnBlock for MI455X (gfx1250): bf16 WMMA (v_wmma_f32_16x16x32_bf16) GEMMs
// + flash-attention, f32 accumulation. All tiles sized to stay < 256 VGPRs
// per lane (the directly addressable VGPR window) so nothing spills.
// ---------------------------------------------------------------------------

typedef __attribute__((ext_vector_type(16))) __bf16 v16bf;
typedef __attribute__((ext_vector_type(8)))  __bf16 v8bf;
typedef __attribute__((ext_vector_type(8)))  float  v8f;

#define B_   4
#define C_   512
#define HW_  4096
#define NG_  32
#define CG_  16   // channels per group

__device__ __forceinline__ uint16_t f2bf(float f) {
  union { float f; uint32_t u; } v; v.f = f;
  uint32_t u = v.u + 0x7FFFu + ((v.u >> 16) & 1u);   // round-to-nearest-even
  return (uint16_t)(u >> 16);
}

__device__ __forceinline__ v16bf combine8(v8bf lo, v8bf hi) {
  return __builtin_shufflevector(lo, hi, 0,1,2,3,4,5,6,7,8,9,10,11,12,13,14,15);
}

__device__ __forceinline__ v8f wmma_bf16(v16bf a, v16bf b, v8f c) {
  // D = A(16x32) * B(32x16) + C, f32 accumulate
  return __builtin_amdgcn_wmma_f32_16x16x32_bf16(false, a, false, b,
                                                 (short)0, c, false, false);
}

// ---------------------------------------------------------------------------
// Kernel 1: GroupNorm. One block per (batch, group). Writes h normalized,
// TRANSPOSED to [B][HW][C] bf16 so GEMM B-fragments are contiguous-K loads.
// ---------------------------------------------------------------------------
__global__ void __launch_bounds__(256)
gn_kernel(const float* __restrict__ x, const float* __restrict__ gw,
          const float* __restrict__ gb, uint16_t* __restrict__ hbT) {
  const int bg = blockIdx.x;              // 0..127
  const int b = bg / NG_, g = bg % NG_;
  const int N = CG_ * HW_;                // 65536 contiguous floats per group
  const float* base = x + ((size_t)b * C_ + (size_t)g * CG_) * HW_;

  float s = 0.f, ss = 0.f;
  for (int i = threadIdx.x; i < N; i += 256) {
    float v = base[i]; s += v; ss += v * v;
  }
  __shared__ float r1[256], r2[256];
  r1[threadIdx.x] = s; r2[threadIdx.x] = ss;
  __syncthreads();
  for (int st = 128; st > 0; st >>= 1) {
    if ((int)threadIdx.x < st) {
      r1[threadIdx.x] += r1[threadIdx.x + st];
      r2[threadIdx.x] += r2[threadIdx.x + st];
    }
    __syncthreads();
  }
  const float mean = r1[0] / (float)N;
  const float var  = r2[0] / (float)N - mean * mean;
  const float rinv = rsqrtf(var + 1e-5f);

  for (int i = threadIdx.x; i < N; i += 256) {
    int cl = i >> 12;              // i / HW_
    int p  = i & (HW_ - 1);
    int c  = g * CG_ + cl;
    float y = (base[i] - mean) * rinv * gw[c] + gb[c];
    hbT[((size_t)b * HW_ + p) * C_ + c] = f2bf(y);
  }
}

// ---------------------------------------------------------------------------
// Kernel 2: f32 -> bf16 convert (weights)
// ---------------------------------------------------------------------------
__global__ void __launch_bounds__(256)
cvt_kernel(const float* __restrict__ src, uint16_t* __restrict__ dst, int n) {
  int i = blockIdx.x * blockDim.x + threadIdx.x;
  if (i < n) dst[i] = f2bf(src[i]);
}

// ---------------------------------------------------------------------------
// Kernel 3: GEMM out[m][n] = sum_k A[m][k] * Bsrc[n][k] + bias[m]
//   A:    bf16 [512][512]   (row major == WMMA A layout source)
//   Bsrc: bf16 [B][4096][512] ([N][K]: lane reads 16 contiguous K == B layout)
// 256 thr = 8 waves; block tile 128(M) x 256(N); wave tile 64x64 (16 accums).
// Inner loop keeps one B fragment live against 4 resident A fragments.
// MODE 0: store bf16 transposed [n][m] with scale (q: C^-0.5 folded in; k: 1)
// MODE 1: store bf16 natural [m][n]              (v)
// MODE 2: store f32 natural + residual           (output projection)
// ---------------------------------------------------------------------------
template<int MODE>
__global__ void __launch_bounds__(256)
gemm512(const uint16_t* __restrict__ A,
        const uint16_t* __restrict__ Bsrc,
        const float* __restrict__ bias,
        uint16_t* __restrict__ outb,
        float* __restrict__ outf,
        const float* __restrict__ resid,
        float scale) {
  const int lane = threadIdx.x & 31;
  const int w    = threadIdx.x >> 5;      // 0..7
  const int ml   = lane & 15;
  const int h    = lane >> 4;             // half-wave id
  const int b    = blockIdx.z;
  const int m0   = blockIdx.y * 128 + (w >> 2) * 64;
  const int n0   = blockIdx.x * 256 + (w & 3) * 64;
  const uint16_t* Bb = Bsrc + (size_t)b * HW_ * C_;

  const v8f vzero = {0.f,0.f,0.f,0.f,0.f,0.f,0.f,0.f};
  v8f acc[4][4];
  #pragma unroll
  for (int i = 0; i < 4; i++)
    #pragma unroll
    for (int j = 0; j < 4; j++) acc[i][j] = vzero;

  for (int k0 = 0; k0 < C_; k0 += 32) {
    v16bf af[4];
    #pragma unroll
    for (int mt = 0; mt < 4; mt++) {
      // A fragment, contiguous k order (same as B): lanes 0-15 hold K = 0..15, lanes 16-31 hold K = 16..31
      const uint16_t* rp = A + (size_t)(m0 + mt * 16 + ml) * C_ + k0;
      v8bf lo = *(const v8bf*)(rp + h * 16);
      v8bf hi = *(const v8bf*)(rp + h * 16 + 8);
      af[mt] = combine8(lo, hi);
    }
    #pragma unroll
    for (int nt = 0; nt < 4; nt++) {
      // B layout: lane = column n, 16 contiguous K per lane (split per half)
      const uint16_t* rp = Bb + (size_t)(n0 + nt * 16 + ml) * C_ + k0 + h * 16;
      v16bf bfr = *(const v16bf*)rp;
      #pragma unroll
      for (int mt = 0; mt < 4; mt++)
        acc[mt][nt] = wmma_bf16(af[mt], bfr, acc[mt][nt]);
    }
  }

  // D layout: VGPR r -> row (r + 8h), lane%16 -> column
  #pragma unroll
  for (int mt = 0; mt < 4; mt++) {
    #pragma unroll
    for (int nt = 0; nt < 4; nt++) {
      #pragma unroll
      for (int r = 0; r < 8; r++) {
        int m = m0 + mt * 16 + r + 8 * h;
        int n = n0 + nt * 16 + ml;
        float val = acc[mt][nt][r] + bias[m];
        if (MODE == 0) {
          outb[((size_t)b * HW_ + n) * C_ + m] = f2bf(val * scale);
        } else if (MODE == 1) {
          outb[((size_t)b * C_ + m) * HW_ + n] = f2bf(val);
        } else {
          size_t idx = ((size_t)b * C_ + m) * HW_ + n;
          outf[idx] = resid[idx] + val;
        }
      }
    }
  }
}

// ---------------------------------------------------------------------------
// Kernel 4: flash attention, restructured for <256 VGPRs/lane.
//   qT [B][HW][C] bf16 (C^-0.5 folded in), kT [B][HW][C] bf16,
//   v  [B][C][HW] bf16  ->  oT [B][HW][C] bf16
// Block = 256 thr = 8 waves covering a 64-query tile:
//   row-group mw = w>>1 (16 query rows), channel-half cw = w&1 (256 channels).
// The two waves of a row-group redundantly compute the identical S[16x32]
// tile + online softmax (no cross-wave reduction), then each accumulates its
// own 256-channel half of P x V^T:  acc[16] = 128 VGPRs.
// ---------------------------------------------------------------------------
__global__ void __launch_bounds__(256)
flash_kernel(const uint16_t* __restrict__ qT,
             const uint16_t* __restrict__ kT,
             const uint16_t* __restrict__ vN,
             uint16_t* __restrict__ oT) {
  const int lane = threadIdx.x & 31;
  const int w    = threadIdx.x >> 5;
  const int ml   = lane & 15;
  const int h    = lane >> 4;
  const int b    = blockIdx.y;
  const int mw   = w >> 1;                      // row group 0..3
  const int cw   = w & 1;                       // channel half 0..1
  const int i0   = blockIdx.x * 64 + mw * 16;   // this wave's query rows
  const int c0   = cw * 256;                    // this wave's channel base

  __shared__ __align__(32) uint16_t Pbuf[8][16][32];  // per-wave P tile

  const uint16_t* qrow = qT + ((size_t)b * HW_ + i0 + ml) * C_;

  const v8f vzero = {0.f,0.f,0.f,0.f,0.f,0.f,0.f,0.f};
  v8f acc[16];
  #pragma unroll
  for (int ct = 0; ct < 16; ct++) acc[ct] = vzero;
  float m_r[8], l_r[8];
  #pragma unroll
  for (int r = 0; r < 8; r++) { m_r[r] = -1e30f; l_r[r] = 0.f; }

  for (int kb = 0; kb < HW_ / 32; kb++) {
    const int j0 = kb * 32;
    const uint16_t* k0p = kT + ((size_t)b * HW_ + j0 + ml) * C_;
    const uint16_t* k1p = kT + ((size_t)b * HW_ + j0 + 16 + ml) * C_;
    if (kb + 1 < HW_ / 32)
      __builtin_prefetch(kT + ((size_t)b * HW_ + j0 + 32 + ml) * C_, 0, 0);

    v8f s0 = vzero, s1 = vzero;
    #pragma unroll
    for (int kc = 0; kc < 16; kc++) {
      v8bf qlo = *(const v8bf*)(qrow + kc * 32 + h * 16);
      v8bf qhi = *(const v8bf*)(qrow + kc * 32 + h * 16 + 8);
      v16bf a  = combine8(qlo, qhi);
      v16bf b0 = *(const v16bf*)(k0p + kc * 32 + h * 16);
      v16bf b1 = *(const v16bf*)(k1p + kc * 32 + h * 16);
      s0 = wmma_bf16(a, b0, s0);
      s1 = wmma_bf16(a, b1, s1);
    }

    // Online softmax. Row (r + 8h) lives across the 16 lanes of half-wave h,
    // so xor-shuffles with masks 1/2/4/8 reduce exactly over one row.
    float alpha[8];
    #pragma unroll
    for (int r = 0; r < 8; r++) {
      float mx = fmaxf(s0[r], s1[r]);
      mx = fmaxf(mx, __shfl_xor(mx, 1));
      mx = fmaxf(mx, __shfl_xor(mx, 2));
      mx = fmaxf(mx, __shfl_xor(mx, 4));
      mx = fmaxf(mx, __shfl_xor(mx, 8));
      float mn = fmaxf(m_r[r], mx);
      alpha[r] = __expf(m_r[r] - mn);
      float p0 = __expf(s0[r] - mn);
      float p1 = __expf(s1[r] - mn);
      float rs = p0 + p1;
      rs += __shfl_xor(rs, 1);
      rs += __shfl_xor(rs, 2);
      rs += __shfl_xor(rs, 4);
      rs += __shfl_xor(rs, 8);
      l_r[r] = l_r[r] * alpha[r] + rs;
      m_r[r] = mn;
      s0[r] = p0; s1[r] = p1;
    }
    #pragma unroll
    for (int ct = 0; ct < 16; ct++)
      #pragma unroll
      for (int r = 0; r < 8; r++) acc[ct][r] *= alpha[r];

    // D-layout P -> LDS (own wave's region) -> re-read in A-layout (bf16)
    #pragma unroll
    for (int r = 0; r < 8; r++) {
      Pbuf[w][r + 8 * h][ml]      = f2bf(s0[r]);
      Pbuf[w][r + 8 * h][16 + ml] = f2bf(s1[r]);
    }
    __syncthreads();
    v8bf plo = *(const v8bf*)&Pbuf[w][ml][h * 16];
    v8bf phi = *(const v8bf*)&Pbuf[w][ml][h * 16 + 8];
    v16bf aP = combine8(plo, phi);

    // acc[i][c] += P[i][j] * v[c][j]; v natural [C][HW] is B layout for K=j
    const uint16_t* vbase = vN + (size_t)b * C_ * HW_ + j0 + h * 16;
    #pragma unroll
    for (int ct = 0; ct < 16; ct++) {
      v16bf bv = *(const v16bf*)(vbase + (size_t)(c0 + ct * 16 + ml) * HW_);
      acc[ct] = wmma_bf16(aP, bv, acc[ct]);
    }
    __syncthreads();
  }

  float inv[8];
  #pragma unroll
  for (int r = 0; r < 8; r++) inv[r] = 1.0f / l_r[r];
  #pragma unroll
  for (int ct = 0; ct < 16; ct++)
    #pragma unroll
    for (int r = 0; r < 8; r++) {
      int i = i0 + r + 8 * h;
      oT[((size_t)b * HW_ + i) * C_ + c0 + ct * 16 + ml] =
          f2bf(acc[ct][r] * inv[r]);
    }
}

// ---------------------------------------------------------------------------
extern "C" void kernel_launch(void* const* d_in, const int* in_sizes, int n_in,
                              void* d_out, int out_size, void* d_ws, size_t ws_size,
                              hipStream_t stream) {
  (void)in_sizes; (void)n_in; (void)out_size; (void)ws_size;
  const float* x    = (const float*)d_in[0];
  const float* gn_w = (const float*)d_in[1];
  const float* gn_b = (const float*)d_in[2];
  const float* wq   = (const float*)d_in[3];
  const float* bq   = (const float*)d_in[4];
  const float* wk   = (const float*)d_in[5];
  const float* bk   = (const float*)d_in[6];
  const float* wv   = (const float*)d_in[7];
  const float* bv   = (const float*)d_in[8];
  const float* wo   = (const float*)d_in[9];
  const float* bo   = (const float*)d_in[10];
  float* out = (float*)d_out;

  char* ws = (char*)d_ws;
  size_t off = 0;
  const size_t HT = (size_t)B_ * HW_ * C_ * sizeof(uint16_t);  // 16 MB each
  uint16_t* hbT = (uint16_t*)(ws + off); off += HT;   // GN(x) transposed [B][HW][C]
  uint16_t* qT  = (uint16_t*)(ws + off); off += HT;   // q scaled, [B][HW][C]
  uint16_t* kTt = (uint16_t*)(ws + off); off += HT;   // k, [B][HW][C]
  uint16_t* vNb = (uint16_t*)(ws + off); off += HT;   // v, [B][C][HW]
  uint16_t* oTT = (uint16_t*)(ws + off); off += HT;   // attn out, [B][HW][C]
  const size_t WT = (size_t)C_ * C_ * sizeof(uint16_t);
  uint16_t* wqb = (uint16_t*)(ws + off); off += WT;
  uint16_t* wkb = (uint16_t*)(ws + off); off += WT;
  uint16_t* wvb = (uint16_t*)(ws + off); off += WT;
  uint16_t* wob = (uint16_t*)(ws + off); off += WT;

  const int wn = C_ * C_;
  cvt_kernel<<<dim3((wn + 255) / 256), 256, 0, stream>>>(wq, wqb, wn);
  cvt_kernel<<<dim3((wn + 255) / 256), 256, 0, stream>>>(wk, wkb, wn);
  cvt_kernel<<<dim3((wn + 255) / 256), 256, 0, stream>>>(wv, wvb, wn);
  cvt_kernel<<<dim3((wn + 255) / 256), 256, 0, stream>>>(wo, wob, wn);

  gn_kernel<<<dim3(B_ * NG_), 256, 0, stream>>>(x, gn_w, gn_b, hbT);

  dim3 gg(HW_ / 256, C_ / 128, B_);               // (16, 4, 4)
  const float scale = 0.04419417382415922f;       // 512^-0.5 folded into q
  gemm512<0><<<gg, 256, 0, stream>>>(wqb, hbT, bq, qT,  nullptr, nullptr, scale);
  gemm512<0><<<gg, 256, 0, stream>>>(wkb, hbT, bk, kTt, nullptr, nullptr, 1.0f);
  gemm512<1><<<gg, 256, 0, stream>>>(wvb, hbT, bv, vNb, nullptr, nullptr, 1.0f);

  flash_kernel<<<dim3(HW_ / 64, B_), 256, 0, stream>>>(qT, kTt, vNb, oTT);

  gemm512<2><<<gg, 256, 0, stream>>>(wob, oTT, bo, nullptr, out, x, 1.0f);
}
